// BiologicallyRealisticRNN_20220706030447
// MI455X (gfx1250) — compile-verified, matrix operand order checked by reading, not run on hardware
//
#include <hip/hip_runtime.h>
#include <hip/hip_bf16.h>

// ---------------------------------------------------------------------------
// Biologically-realistic RNN, persistent-kernel formulation for gfx1250.
// Phase 1 (parallel, bandwidth-bound): x f32 -> bf16 into workspace.
// Phase 2 (persistent, latency-bound): 512 sequential steps, each
//   [64x1024] += [64x1024]_bf16 @ w_rec^T_bf16  (+ x_t @ w_in^T)
// via v_wmma_f32_16x16x32_bf16; f32 state lives in registers (C/D layout);
// masked bf16 w_rec / w_in slices resident in 320KB WGP LDS for all 512 steps;
// one device-wide atomic barrier per step; bf16 ping-pong output in ws.
// ---------------------------------------------------------------------------

#define TT 512
#define BB 64
#define II 256
#define HH 1024
#define NWG 64            // 4 b-tiles x 16 h-tiles
#define WREC_STRIDE 1032  // 1024 + 8 bf16 pad -> 2064B row stride (bank-rotated)
#define WIN_STRIDE  264   // 256 + 8

typedef __attribute__((ext_vector_type(16))) __bf16 v16bf;
typedef __attribute__((ext_vector_type(8)))  __bf16 v8bf;
typedef __attribute__((ext_vector_type(8)))  float  v8f;

__device__ __forceinline__ __bf16 f2bf(float f) {
    unsigned u = __builtin_bit_cast(unsigned, f);
    u += 0x7FFFu + ((u >> 16) & 1u);                 // round-to-nearest-even
    unsigned short s = (unsigned short)(u >> 16);
    return __builtin_bit_cast(__bf16, s);
}

__device__ __forceinline__ v16bf cat8(v8bf lo, v8bf hi) {
    return __builtin_shufflevector(lo, hi,
        0,1,2,3,4,5,6,7,8,9,10,11,12,13,14,15);
}

// zero the bf16 ping-pong output buffers and the grid-barrier counter
__global__ void init_ws_kernel(unsigned* w, int n) {
    int i = blockIdx.x * blockDim.x + threadIdx.x;
    if (i < n) w[i] = 0u;
}

// phase 1: convert x [T*B*I] f32 -> bf16 (8 elements per thread)
__global__ void x_to_bf16_kernel(const float* __restrict__ x,
                                 __bf16* __restrict__ xbf, int n8) {
    int i = blockIdx.x * blockDim.x + threadIdx.x;
    if (i >= n8) return;
    const float* src = x + (size_t)i * 8;
    float4 f0 = *(const float4*)(src);
    float4 f1 = *(const float4*)(src + 4);
    v8bf v;
    v[0] = f2bf(f0.x); v[1] = f2bf(f0.y); v[2] = f2bf(f0.z); v[3] = f2bf(f0.w);
    v[4] = f2bf(f1.x); v[5] = f2bf(f1.y); v[6] = f2bf(f1.z); v[7] = f2bf(f1.w);
    *(v8bf*)(xbf + (size_t)i * 8) = v;
}

template <bool XPRE>
__global__ __launch_bounds__(128, 1)
void rnn_persistent_kernel(const float* __restrict__ x,
                           const __bf16* __restrict__ xbf,
                           const float* __restrict__ w_in,
                           const float* __restrict__ b_in,
                           const float* __restrict__ w_h2h,
                           const float* __restrict__ b_h2h,
                           const float* __restrict__ dale,
                           const float* __restrict__ sparse,
                           float* __restrict__ out,
                           __bf16* __restrict__ buf0,
                           __bf16* __restrict__ buf1,
                           unsigned* __restrict__ cnt) {
    extern __shared__ char smem[];
    __bf16* lds_wrec = (__bf16*)smem;                       // 64 x WREC_STRIDE
    __bf16* lds_win  = lds_wrec + 64 * WREC_STRIDE;         // 64 x WIN_STRIDE
    float*  lds_bias = (float*)(lds_win + 64 * WIN_STRIDE); // 64

    const int wg    = blockIdx.x;
    const int b0    = (wg & 3) * 16;       // batch-row tile base
    const int h0    = (wg >> 2) * 64;      // hidden-col tile base (64 cols / WG)
    const int lane  = threadIdx.x & 31;
    const int wave  = threadIdx.x >> 5;    // 4 waves, one 16x16 strip each
    const int khalf = lane >> 4;           // K-half select per A/B VGPR layout
    const int mrow  = lane & 15;           // row (A) / col (B,C,D) index
    const int hs    = h0 + wave * 16;

    // ---- one-time: build masked bf16 w_rec slice + w_in slice in LDS -------
    for (int idx = threadIdx.x; idx < 64 * HH; idx += 128) {
        int r = idx >> 10, k = idx & (HH - 1);
        size_t g = (size_t)(h0 + r) * HH + k;
        float w = fmaxf(w_h2h[g], 0.0f) * dale[k] * sparse[g];
        lds_wrec[r * WREC_STRIDE + k] = f2bf(w);
    }
    for (int idx = threadIdx.x; idx < 64 * II; idx += 128) {
        int r = idx >> 8, k = idx & (II - 1);
        lds_win[r * WIN_STRIDE + k] = f2bf(w_in[(size_t)(h0 + r) * II + k]);
    }
    for (int idx = threadIdx.x; idx < 64; idx += 128)
        lds_bias[idx] = b_in[h0 + idx] + b_h2h[h0 + idx];
    __syncthreads();

    const __bf16* wrecrow = lds_wrec + (wave * 16 + mrow) * WREC_STRIDE;
    const __bf16* winrow  = lds_win  + (wave * 16 + mrow) * WIN_STRIDE;
    const float   biasv   = lds_bias[wave * 16 + mrow];

    v8f st = {};  // f32 state tile (C/D layout: element r -> row b0+r+8*khalf)

    float* ys     = out;                          // [T,B,H]
    float* st_out = out + (size_t)TT * BB * HH;   // final state [B,H]
    float* fo_out = st_out + (size_t)BB * HH;     // final output [B,H]

    for (int t = 0; t < TT; ++t) {
        const __bf16* bufPrev = (t & 1) ? buf1 : buf0;
        __bf16*       bufNext = (t & 1) ? buf0 : buf1;

        v8f c = {};

        // ---- input projection: 8x wmma over I=256 ------------------------
        if constexpr (XPRE) {
            const __bf16* xrow = xbf + ((size_t)t * BB + b0 + mrow) * II;
            #pragma unroll
            for (int k0 = 0; k0 < II; k0 += 32) {
                v16bf a = cat8(*(const v8bf*)(xrow + k0 + 8 * khalf),
                               *(const v8bf*)(xrow + k0 + 16 + 8 * khalf));
                v16bf b = cat8(*(const v8bf*)(winrow + k0 + 8 * khalf),
                               *(const v8bf*)(winrow + k0 + 16 + 8 * khalf));
                c = __builtin_amdgcn_wmma_f32_16x16x32_bf16(
                        false, a, false, b, (short)0, c, false, false);
            }
        } else {
            const float* xrow = x + ((size_t)t * BB + b0 + mrow) * II;
            #pragma unroll
            for (int k0 = 0; k0 < II; k0 += 32) {
                alignas(16) float tmp[16];
                *(float4*)(tmp + 0)  = *(const float4*)(xrow + k0 + 8 * khalf);
                *(float4*)(tmp + 4)  = *(const float4*)(xrow + k0 + 8 * khalf + 4);
                *(float4*)(tmp + 8)  = *(const float4*)(xrow + k0 + 16 + 8 * khalf);
                *(float4*)(tmp + 12) = *(const float4*)(xrow + k0 + 16 + 8 * khalf + 4);
                v16bf a;
                #pragma unroll
                for (int e = 0; e < 16; ++e) a[e] = f2bf(tmp[e]);
                v16bf b = cat8(*(const v8bf*)(winrow + k0 + 8 * khalf),
                               *(const v8bf*)(winrow + k0 + 16 + 8 * khalf));
                c = __builtin_amdgcn_wmma_f32_16x16x32_bf16(
                        false, a, false, b, (short)0, c, false, false);
            }
        }

        // ---- recurrent GEMM: 32x wmma over H=1024 ------------------------
        const __bf16* arow = bufPrev + (size_t)(b0 + mrow) * HH;
        #pragma unroll 8
        for (int k0 = 0; k0 < HH; k0 += 32) {
            v16bf a = cat8(*(const v8bf*)(arow + k0 + 8 * khalf),
                           *(const v8bf*)(arow + k0 + 16 + 8 * khalf));
            v16bf b = cat8(*(const v8bf*)(wrecrow + k0 + 8 * khalf),
                           *(const v8bf*)(wrecrow + k0 + 16 + 8 * khalf));
            c = __builtin_amdgcn_wmma_f32_16x16x32_bf16(
                    false, a, false, b, (short)0, c, false, false);
        }

        // ---- leaky-integrator update, ReLU, stores -----------------------
        #pragma unroll
        for (int r = 0; r < 8; ++r) {
            int   brow  = b0 + r + 8 * khalf;
            float total = c[r] + biasv;
            st[r] = st[r] * 0.8f + total * 0.2f;
            float o = fmaxf(st[r], 0.0f);
            __builtin_nontemporal_store(
                o, ys + ((size_t)t * BB + brow) * HH + hs + mrow);
            bufNext[(size_t)brow * HH + hs + mrow] = f2bf(o);
        }

        // ---- device-wide step barrier (release -> arrive -> acquire) -----
        __threadfence();
        __syncthreads();
        if (threadIdx.x == 0) {
            __hip_atomic_fetch_add(cnt, 1u, __ATOMIC_RELEASE,
                                   __HIP_MEMORY_SCOPE_AGENT);
            unsigned target = (unsigned)NWG * (unsigned)(t + 1);
            while (__hip_atomic_load(cnt, __ATOMIC_ACQUIRE,
                                     __HIP_MEMORY_SCOPE_AGENT) < target)
                __builtin_amdgcn_s_sleep(2);
        }
        __syncthreads();
        __threadfence();
    }

    // ---- final state / output ------------------------------------------
    #pragma unroll
    for (int r = 0; r < 8; ++r) {
        int brow = b0 + r + 8 * khalf;
        size_t idx = (size_t)brow * HH + hs + mrow;
        st_out[idx] = st[r];
        fo_out[idx] = fmaxf(st[r], 0.0f);
    }
}

extern "C" void kernel_launch(void* const* d_in, const int* in_sizes, int n_in,
                              void* d_out, int out_size, void* d_ws, size_t ws_size,
                              hipStream_t stream) {
    const float* x      = (const float*)d_in[0];
    const float* w_in   = (const float*)d_in[1];
    const float* b_in   = (const float*)d_in[2];
    const float* w_h2h  = (const float*)d_in[3];
    const float* b_h2h  = (const float*)d_in[4];
    const float* dale   = (const float*)d_in[5];
    const float* sparse = (const float*)d_in[6];
    float* out = (float*)d_out;

    // ws layout: buf0 | buf1 (B*H bf16 each) | counter block | optional x_bf16
    const size_t buf_bytes = (size_t)BB * HH * sizeof(__bf16);   // 128 KB
    const size_t xbf_off   = 2 * buf_bytes + 256;                // counter block
    const size_t x_bytes   = (size_t)TT * BB * II * sizeof(__bf16); // 16.8 MB

    __bf16* buf0 = (__bf16*)d_ws;
    __bf16* buf1 = (__bf16*)((char*)d_ws + buf_bytes);
    unsigned* cnt = (unsigned*)((char*)d_ws + 2 * buf_bytes);
    __bf16* xbf = (__bf16*)((char*)d_ws + xbf_off);

    // zero ping-pong buffers + counter every call (harness doesn't re-init ws)
    int init_words = (int)(2 * buf_bytes / 4) + 1;
    init_ws_kernel<<<(init_words + 255) / 256, 256, 0, stream>>>(
        (unsigned*)d_ws, init_words);

    size_t lds_bytes = (size_t)64 * WREC_STRIDE * sizeof(__bf16)
                     + (size_t)64 * WIN_STRIDE * sizeof(__bf16)
                     + 64 * sizeof(float);

    const bool xpre = (ws_size >= xbf_off + x_bytes);
    if (xpre) {
        int n8 = (int)((size_t)TT * BB * II / 8);
        x_to_bf16_kernel<<<(n8 + 255) / 256, 256, 0, stream>>>(x, xbf, n8);
        rnn_persistent_kernel<true><<<NWG, 128, lds_bytes, stream>>>(
            x, xbf, w_in, b_in, w_h2h, b_h2h, dale, sparse, out, buf0, buf1, cnt);
    } else {
        rnn_persistent_kernel<false><<<NWG, 128, lds_bytes, stream>>>(
            x, xbf, w_in, b_in, w_h2h, b_h2h, dale, sparse, out, buf0, buf1, cnt);
    }
}
